// LinkPredictionGNN_33749853012397
// MI455X (gfx1250) — compile-verified, matrix operand order checked by reading, not run on hardware
//
#include <hip/hip_runtime.h>
#include <cstdint>
#include <cstddef>

// ---------------- types / helpers ----------------
typedef __attribute__((ext_vector_type(16))) _Float16 v16h;
typedef __attribute__((ext_vector_type(8)))  _Float16 v8h;
typedef __attribute__((ext_vector_type(8)))  float    v8f;
typedef __attribute__((ext_vector_type(4)))  _Float16 v4h;

#define F_IN 32
#define F_E  8
#define HDIM 64
#define NEG_SLOPE 0.2f

static __device__ __forceinline__ v16h cat8(v8h lo, v8h hi) {
  return __builtin_shufflevector(lo, hi, 0, 1, 2, 3, 4, 5, 6, 7,
                                         8, 9, 10, 11, 12, 13, 14, 15);
}

// A/B-fragment (16-bit, 16xK tile, K-offset kofs) for one lane: two contiguous
// 8-half runs at K = kofs+half*8 and K = kofs+16+half*8 (ISA 7.12.2 layout).
static __device__ __forceinline__ v16h load_frag_f16(const _Float16* __restrict__ rowp,
                                                     int half, int kofs) {
  v8h lo = *(const v8h*)(rowp + kofs + half * 8);
  v8h hi = *(const v8h*)(rowp + kofs + 16 + half * 8);
  return cat8(lo, hi);
}

static __device__ __forceinline__ void atomic_max_float(float* addr, float v) {
  // ordered-bits trick; addr initialized to -inf (0xFF800000)
  if (v >= 0.0f) atomicMax((int*)addr, __float_as_int(v));
  else           atomicMin((unsigned int*)addr, (unsigned int)__float_as_int(v));
}

// ---------------- tiny setup kernels ----------------
__global__ void k_zero_small(float* small) {
  if (threadIdx.x < 64) small[threadIdx.x] = 0.0f;
}

// column sums of edge_attr -> small[0..7]
__global__ void k_sum_eattr(const float* __restrict__ eattr, float* __restrict__ small, int nE) {
  float s[F_E];
#pragma unroll
  for (int k = 0; k < F_E; ++k) s[k] = 0.0f;
  for (int e = blockIdx.x * blockDim.x + threadIdx.x; e < nE; e += gridDim.x * blockDim.x) {
    const float* ea = eattr + (size_t)e * F_E;
#pragma unroll
    for (int k = 0; k < F_E; ++k) s[k] += ea[k];
  }
  __shared__ float red[256];
  int t = threadIdx.x;
#pragma unroll
  for (int k = 0; k < F_E; ++k) {
    red[t] = s[k];
    __syncthreads();
    for (int off = 128; off > 0; off >>= 1) {
      if (t < off) red[t] += red[t + off];
      __syncthreads();
    }
    if (t == 0) atomicAdd(&small[k], red[0]);
    __syncthreads();
  }
}

// ae[k] = sum_j edge_w[k][j]*att_edge[j]  -> small[8..15];  loop_alpha -> small[16]
__global__ void k_ae(const float* __restrict__ edge_w, const float* __restrict__ att_edge,
                     float* __restrict__ small, int nE) {
  int t = threadIdx.x;
  if (t < F_E) {
    float s = 0.0f;
#pragma unroll
    for (int j = 0; j < HDIM; ++j) s += edge_w[t * HDIM + j] * att_edge[j];
    small[8 + t] = s;
  }
  __syncthreads();
  if (t == 0) {
    float la = 0.0f;
#pragma unroll
    for (int k = 0; k < F_E; ++k) la += small[k] * small[8 + k];
    small[16] = la / (float)nE;  // dot(mean_eattr, ae)
  }
}

// ---------------- encoder: h = x @ enc_w + enc_b (WMMA f16) ----------------
__global__ void k_encoder(const float* __restrict__ x, const float* __restrict__ W,
                          const float* __restrict__ bias, float* __restrict__ h,
                          _Float16* __restrict__ hh, int nN) {
  // weights stored TRANSPOSED: sWt[col*32 + k] so B fragments are contiguous b128 loads
  __shared__ _Float16 sWt[HDIM * F_IN];
  __shared__ float sB[HDIM];
  int t = threadIdx.x;
  for (int i = t; i < F_IN * HDIM; i += 256) {
    int k = i >> 6, col = i & 63;                  // i = k*64 + col
    sWt[col * F_IN + k] = (_Float16)W[i];
  }
  if (t < HDIM) sB[t] = bias[t];
  __syncthreads();

  int wave = t >> 5, lane = t & 31;
  int nT = (nN + 15) >> 4;
  // wave-uniform tile index -> force into SGPR so bounds checks are scalar branches
  int tile = __builtin_amdgcn_readfirstlane(blockIdx.x * 8 + wave);
  if (tile >= nT) return;
  int half = lane >> 4, l15 = lane & 15;

  int row = tile * 16 + l15;
  int rowc = row < nN ? row : nN - 1;
  const float* xr = x + (size_t)rowc * F_IN;
  // A fragment from f32: two 32B runs at K = half*8 and K = 16+half*8
  v16h a;
#pragma unroll
  for (int u = 0; u < 8; ++u) {
    a[u]     = (_Float16)xr[half * 8 + u];
    a[8 + u] = (_Float16)xr[16 + half * 8 + u];
  }

  const bool full = (tile * 16 + 16) <= nN;  // scalar (tile, nN uniform)
#pragma unroll
  for (int nt = 0; nt < 4; ++nt) {
    int col = nt * 16 + l15;
    v16h b = load_frag_f16(&sWt[col * F_IN], half, 0);
    v8f c = {};
    c = __builtin_amdgcn_wmma_f32_16x16x32_f16(false, a, false, b, (short)0, c, false, false);
    float bb = sB[col];
    if (full) {
#pragma unroll
      for (int r = 0; r < 8; ++r) {
        int orow = tile * 16 + r + half * 8;
        float v = c[r] + bb;
        h[(size_t)orow * HDIM + col]  = v;
        hh[(size_t)orow * HDIM + col] = (_Float16)v;
      }
    } else {
#pragma unroll
      for (int r = 0; r < 8; ++r) {
        int orow = tile * 16 + r + half * 8;
        if (orow < nN) {
          float v = c[r] + bb;
          h[(size_t)orow * HDIM + col]  = v;
          hh[(size_t)orow * HDIM + col] = (_Float16)v;
        }
      }
    }
  }
}

// ---------------- xp = h @ W_l (64x64) via WMMA, K=64 ----------------
__global__ void k_xp(const _Float16* __restrict__ hh, const float* __restrict__ W,
                     float* __restrict__ xp, int nN) {
  __shared__ _Float16 sWt[HDIM * HDIM];  // transposed: sWt[col*64 + k]
  int t = threadIdx.x;
  for (int i = t; i < HDIM * HDIM; i += 256) {
    int k = i >> 6, col = i & 63;
    sWt[col * HDIM + k] = (_Float16)W[i];
  }
  __syncthreads();

  int wave = t >> 5, lane = t & 31;
  int nT = (nN + 15) >> 4;
  int tile = __builtin_amdgcn_readfirstlane(blockIdx.x * 8 + wave);
  if (tile >= nT) return;
  int half = lane >> 4, l15 = lane & 15;

  int row = tile * 16 + l15;
  int rowc = row < nN ? row : nN - 1;
  const _Float16* hr = hh + (size_t)rowc * HDIM;

  v8f c[4] = {};
#pragma unroll
  for (int ks = 0; ks < 2; ++ks) {
    v16h a = load_frag_f16(hr, half, ks * 32);
#pragma unroll
    for (int nt = 0; nt < 4; ++nt) {
      int col = nt * 16 + l15;
      v16h b = load_frag_f16(&sWt[col * HDIM], half, ks * 32);
      c[nt] = __builtin_amdgcn_wmma_f32_16x16x32_f16(false, a, false, b, (short)0, c[nt], false, false);
    }
  }

  const bool full = (tile * 16 + 16) <= nN;  // scalar
#pragma unroll
  for (int nt = 0; nt < 4; ++nt) {
    int col = nt * 16 + l15;
    if (full) {
#pragma unroll
      for (int r = 0; r < 8; ++r) {
        int orow = tile * 16 + r + half * 8;
        xp[(size_t)orow * HDIM + col] = c[nt][r];
      }
    } else {
#pragma unroll
      for (int r = 0; r < 8; ++r) {
        int orow = tile * 16 + r + half * 8;
        if (orow < nN) xp[(size_t)orow * HDIM + col] = c[nt][r];
      }
    }
  }
}

// ---------------- per-node attention scores ----------------
__global__ void k_scores(const float* __restrict__ xp, const float* __restrict__ a_src,
                         const float* __restrict__ a_dst, float* __restrict__ ssrc,
                         float* __restrict__ sdst, int nN) {
  __shared__ float sa[HDIM], sb[HDIM];
  int t = threadIdx.x;
  if (t < HDIM) { sa[t] = a_src[t]; sb[t] = a_dst[t]; }
  __syncthreads();
  int node = blockIdx.x * 256 + t;
  if (node >= nN) return;
  const float* r = xp + (size_t)node * HDIM;
  float s1 = 0.0f, s2 = 0.0f;
#pragma unroll
  for (int i = 0; i < HDIM; ++i) { float v = r[i]; s1 += v * sa[i]; s2 += v * sb[i]; }
  ssrc[node] = s1; sdst[node] = s2;
}

// ---------------- init segment buffers + zero accumulator ----------------
__global__ void k_init(float* __restrict__ h, float* __restrict__ mseg,
                       float* __restrict__ dseg, int nN) {
  int node = blockIdx.x * 256 + threadIdx.x;
  if (node >= nN) return;
  mseg[node] = __int_as_float(0xFF800000);  // -inf
  dseg[node] = 0.0f;
  float4* hp = (float4*)(h + (size_t)node * HDIM);
  float4 z = make_float4(0.f, 0.f, 0.f, 0.f);
#pragma unroll
  for (int i = 0; i < HDIM / 4; ++i) hp[i] = z;
}

// ---------------- alpha + leaky relu + segment max ----------------
__global__ void k_alpha_max(const int* __restrict__ ei, const float* __restrict__ eattr,
                            const float* __restrict__ ssrc, const float* __restrict__ sdst,
                            const float* __restrict__ small, float* __restrict__ alpha,
                            float* __restrict__ mseg, int nE, int nN) {
  int e = blockIdx.x * 256 + threadIdx.x;
  if (e >= nE + nN) return;
  int d; float av;
  if (e < nE) {
    int s = ei[e]; d = ei[nE + e];
    const float* ea = eattr + (size_t)e * F_E;
    float ed = 0.0f;
#pragma unroll
    for (int k = 0; k < F_E; ++k) ed += ea[k] * small[8 + k];
    av = ssrc[s] + sdst[d] + ed;
  } else {
    int i = e - nE; d = i;
    av = ssrc[i] + sdst[i] + small[16];
  }
  av = av > 0.0f ? av : NEG_SLOPE * av;
  alpha[e] = av;
  atomic_max_float(&mseg[d], av);
}

// ---------------- exp + segment sum ----------------
__global__ void k_exp_sum(const int* __restrict__ ei, float* __restrict__ alpha,
                          const float* __restrict__ mseg, float* __restrict__ dseg,
                          int nE, int nN) {
  int e = blockIdx.x * 256 + threadIdx.x;
  if (e >= nE + nN) return;
  int d = (e < nE) ? ei[nE + e] : (e - nE);
  float ex = __expf(alpha[e] - mseg[d]);
  alpha[e] = ex;
  atomicAdd(&dseg[d], ex);
}

// ---------------- weighted scatter-aggregate: h[d] += w * xp[s] ----------------
__global__ void k_agg(const int* __restrict__ ei, const float* __restrict__ alpha,
                      const float* __restrict__ dseg, const float* __restrict__ xp,
                      float* __restrict__ h, int nE, int nN) {
  size_t tid = (size_t)blockIdx.x * 256 + threadIdx.x;
  size_t e = tid >> 4;
  if (e >= (size_t)(nE + nN)) return;
  int lane = (int)(tid & 15);
  int s, d;
  if (e < (size_t)nE) { s = ei[e]; d = ei[(size_t)nE + e]; }
  else { s = d = (int)(e - nE); }
  float w = alpha[e] / (dseg[d] + 1e-16f);
  const float4 v = *(const float4*)(xp + (size_t)s * HDIM + lane * 4);
  float* hr = h + (size_t)d * HDIM + lane * 4;
  atomicAdd(hr + 0, w * v.x);
  atomicAdd(hr + 1, w * v.y);
  atomicAdd(hr + 2, w * v.z);
  atomicAdd(hr + 3, w * v.w);
}

// ---------------- bias + relu + f16 refresh ----------------
__global__ void k_finish(float* __restrict__ h, const float* __restrict__ bias,
                         _Float16* __restrict__ hh, int nN) {
  int tid = blockIdx.x * 256 + threadIdx.x;
  if (tid >= nN * (HDIM / 4)) return;
  int node = tid >> 4, c4 = tid & 15;
  float4 v = *(float4*)(h + (size_t)node * HDIM + c4 * 4);
  const float* b = bias + c4 * 4;
  v.x = fmaxf(v.x + b[0], 0.0f);
  v.y = fmaxf(v.y + b[1], 0.0f);
  v.z = fmaxf(v.z + b[2], 0.0f);
  v.w = fmaxf(v.w + b[3], 0.0f);
  *(float4*)(h + (size_t)node * HDIM + c4 * 4) = v;
  v4h hv; hv[0] = (_Float16)v.x; hv[1] = (_Float16)v.y; hv[2] = (_Float16)v.z; hv[3] = (_Float16)v.w;
  *(v4h*)(hh + (size_t)node * HDIM + c4 * 4) = hv;
}

// ---------------- link predictor: sigmoid(relu([h_s||h_d] @ W1 + b1) @ W2 + b2) ----------------
__global__ void k_linkpred(const int* __restrict__ ei, const _Float16* __restrict__ hh,
                           const float* __restrict__ w1, const float* __restrict__ b1,
                           const float* __restrict__ w2, const float* __restrict__ b2,
                           float* __restrict__ out, int nE) {
  __shared__ _Float16 sWt[HDIM * 2 * HDIM];  // transposed: sWt[col*128 + k], 16 KB
  __shared__ float sB1[HDIM], sW2[HDIM];
  int t = threadIdx.x;
  for (int i = t; i < 2 * HDIM * HDIM; i += 256) {
    int k = i >> 6, col = i & 63;              // w1 is [128][64]
    sWt[col * 2 * HDIM + k] = (_Float16)w1[i];
  }
  if (t < HDIM) { sB1[t] = b1[t]; sW2[t] = w2[t]; }
  __syncthreads();

  int wave = t >> 5, lane = t & 31;
  int nT = (nE + 15) >> 4;
  int tile = __builtin_amdgcn_readfirstlane(blockIdx.x * 8 + wave);
  if (tile >= nT) return;
  int half = lane >> 4, l15 = lane & 15;

  int eb = tile * 16;
  int e = eb + l15;
  int ec = e < nE ? e : nE - 1;
  int s = ei[ec], d = ei[nE + ec];
  const _Float16* srow = hh + (size_t)s * HDIM;
  const _Float16* drow = hh + (size_t)d * HDIM;

  v8f c[4] = {};
#pragma unroll
  for (int ks = 0; ks < 4; ++ks) {
    const _Float16* rp = (ks < 2) ? srow : drow;
    v16h a = load_frag_f16(rp, half, (ks & 1) * 32);
#pragma unroll
    for (int nt = 0; nt < 4; ++nt) {
      int col = nt * 16 + l15;
      v16h b = load_frag_f16(&sWt[col * 2 * HDIM], half, ks * 32);
      c[nt] = __builtin_amdgcn_wmma_f32_16x16x32_f16(false, a, false, b, (short)0, c[nt], false, false);
    }
  }

  // second MLP layer (64 -> 1) folded into per-row shuffle reduction
  float p[8];
#pragma unroll
  for (int r = 0; r < 8; ++r) {
    float acc = 0.0f;
#pragma unroll
    for (int nt = 0; nt < 4; ++nt) {
      int col = nt * 16 + l15;
      float z = fmaxf(c[nt][r] + sB1[col], 0.0f);
      acc += z * sW2[col];
    }
    p[r] = acc;
  }
#pragma unroll
  for (int off = 8; off > 0; off >>= 1) {
#pragma unroll
    for (int r = 0; r < 8; ++r) p[r] += __shfl_xor(p[r], off, 16);
  }
  if (l15 == 0) {
    float bb = b2[0];
#pragma unroll
    for (int r = 0; r < 8; ++r) {
      int eo = eb + r + half * 8;
      if (eo < nE) {
        float logit = p[r] + bb;
        out[eo] = 1.0f / (1.0f + __expf(-logit));
      }
    }
  }
}

// ---------------- launcher ----------------
extern "C" void kernel_launch(void* const* d_in, const int* in_sizes, int n_in,
                              void* d_out, int out_size, void* d_ws, size_t ws_size,
                              hipStream_t stream) {
  const float* x        = (const float*)d_in[0];
  const int*   ei       = (const int*)  d_in[1];
  const float* eattr    = (const float*)d_in[2];
  const float* enc_w    = (const float*)d_in[3];
  const float* enc_b    = (const float*)d_in[4];
  const float* gat_w    = (const float*)d_in[5];
  const float* att_src  = (const float*)d_in[6];
  const float* att_dst  = (const float*)d_in[7];
  const float* edge_w   = (const float*)d_in[8];
  const float* att_edge = (const float*)d_in[9];
  const float* gat_b    = (const float*)d_in[10];
  const float* lp_w1    = (const float*)d_in[11];
  const float* lp_b1    = (const float*)d_in[12];
  const float* lp_w2    = (const float*)d_in[13];
  const float* lp_b2    = (const float*)d_in[14];
  float* out = (float*)d_out;

  const int nN = in_sizes[0] / F_IN;   // 100000
  const int nE = in_sizes[2] / F_E;    // 1600000
  const int L  = 2;

  size_t off = 0;
  char* wsb = (char*)d_ws;
  auto ws_alloc = [&](size_t bytes) -> void* {
    void* p = (void*)(wsb + off);
    off = (off + bytes + 255) & ~(size_t)255;
    return p;
  };
  float*    h     = (float*)   ws_alloc((size_t)nN * HDIM * 4);
  _Float16* hh    = (_Float16*)ws_alloc((size_t)nN * HDIM * 2);
  float*    xp    = (float*)   ws_alloc((size_t)nN * HDIM * 4);
  float*    ssrc  = (float*)   ws_alloc((size_t)nN * 4);
  float*    sdst  = (float*)   ws_alloc((size_t)nN * 4);
  float*    mseg  = (float*)   ws_alloc((size_t)nN * 4);
  float*    dseg  = (float*)   ws_alloc((size_t)nN * 4);
  float*    alpha = (float*)   ws_alloc((size_t)(nE + nN) * 4);
  float*    small = (float*)   ws_alloc(256);

  const unsigned encBlk = (unsigned)(((nN + 15) / 16 + 7) / 8);
  const unsigned ndBlk  = (unsigned)((nN + 255) / 256);
  const unsigned edBlk  = (unsigned)((nE + nN + 255) / 256);
  const unsigned agBlk  = (unsigned)(((size_t)(nE + nN) * 16 + 255) / 256);
  const unsigned fiBlk  = (unsigned)((nN * (HDIM / 4) + 255) / 256);
  const unsigned lpBlk  = (unsigned)(((nE + 15) / 16 + 7) / 8);

  k_zero_small<<<1, 64, 0, stream>>>(small);
  k_sum_eattr<<<512, 256, 0, stream>>>(eattr, small, nE);
  k_encoder<<<encBlk, 256, 0, stream>>>(x, enc_w, enc_b, h, hh, nN);

  for (int l = 0; l < L; ++l) {
    k_ae<<<1, 64, 0, stream>>>(edge_w + (size_t)l * F_E * HDIM, att_edge + (size_t)l * HDIM, small, nE);
    k_xp<<<encBlk, 256, 0, stream>>>(hh, gat_w + (size_t)l * HDIM * HDIM, xp, nN);
    k_scores<<<ndBlk, 256, 0, stream>>>(xp, att_src + (size_t)l * HDIM, att_dst + (size_t)l * HDIM,
                                        ssrc, sdst, nN);
    k_init<<<ndBlk, 256, 0, stream>>>(h, mseg, dseg, nN);
    k_alpha_max<<<edBlk, 256, 0, stream>>>(ei, eattr, ssrc, sdst, small, alpha, mseg, nE, nN);
    k_exp_sum<<<edBlk, 256, 0, stream>>>(ei, alpha, mseg, dseg, nE, nN);
    k_agg<<<agBlk, 256, 0, stream>>>(ei, alpha, dseg, xp, h, nE, nN);
    k_finish<<<fiBlk, 256, 0, stream>>>(h, gat_b + (size_t)l * HDIM, hh, nN);
  }

  k_linkpred<<<lpBlk, 256, 0, stream>>>(ei, hh, lp_w1, lp_b1, lp_w2, lp_b2, out, nE);
}
